// QueryFeaturesEmbeddings_6305011990954
// MI455X (gfx1250) — matrix operand order repaired and checked by reading, not compiled, not run on hardware
//
#include <hip/hip_runtime.h>

typedef __bf16 bf16_t;
typedef bf16_t v16bf __attribute__((ext_vector_type(16)));
typedef float  v8f   __attribute__((ext_vector_type(8)));

union AF { v16bf v; bf16_t e[16]; };
union CF { v8f v; float f[8]; };

// ---------------- constants ----------------
constexpr int Bc = 64, Qc = 50, Gc = Bc * Qc;   // 3200 graphs
constexpr int Nn = 32, Dd = 128, Hh = 4, Ee = 64;
constexpr int HD = Hh * Dd;                      // 512
constexpr int LDH = 516;                         // padded f32 stride for [32][512] LDS tiles

// ---------------- LDS layout (bytes) ----------------
constexpr int OFF_XL   = 0;        // float[32*516]   66048   (later: hout f32[32*128])
constexpr int OFF_XR   = 66048;    // float[32*516]   66048   (later: agg)
constexpr int OFF_WE   = 132096;   // float[8*512]    16384
constexpr int OFF_ATT  = 148480;   // float[512]       2048
constexpr int OFF_XBF  = 150528;   // bf16[32*128]     8192   (later: gout bf16)
constexpr int OFF_H1BF = 158720;   // bf16[32*128]     8192
constexpr int OFF_EEMB = 166912;   // float[64*8]      2048
constexpr int OFF_SRC  = 168960;   // int[64]           256
constexpr int OFF_DST  = 169216;   // int[64]           256
constexpr int OFF_EMK  = 169472;   // float[64]         256
constexpr int OFF_LOOP = 169728;   // float[32*8]      1024
constexpr int OFF_ECNT = 170752;   // float[32]         128
constexpr int OFF_LOG  = 170880;   // float[64*4]      1024
constexpr int OFF_LGS  = 171904;   // float[32*4]       512
constexpr int OFF_M    = 172416;   // float[32*4]       512
constexpr int OFF_DEN  = 172928;   // float[32*4]       512
constexpr int OFF_PE   = 173440;   // float[64*4]      1024
constexpr int OFF_ALS  = 174464;   // float[32*4]       512
constexpr int OFF_RED  = 174976;   // float[512]       2048
constexpr int OFF_POOL = 177024;   // float[128]        512
constexpr int OFF_STAT = 177536;   // float[16]          64
constexpr int SMEM_BYTES = 177600;

// 16x16 f32 = A(16x128 bf16) x Bt(16 cols x 128 K, col-major rows) via 4 chained WMMAs
__device__ inline v8f wmma_tile128(const bf16_t* __restrict__ A, int lda, int m0,
                                   const bf16_t* __restrict__ Bt, int ldb, int n0,
                                   int lane) {
  const int r = lane & 15, half = lane >> 4;
  v8f acc = {0.f, 0.f, 0.f, 0.f, 0.f, 0.f, 0.f, 0.f};
#pragma unroll
  for (int kb = 0; kb < 128; kb += 32) {
    AF a, b;
    const bf16_t* ap = A + (size_t)(m0 + r) * lda + kb;
#pragma unroll
    for (int i = 0; i < 8; ++i) {
      a.e[i]     = ap[16 * half + i];
      a.e[8 + i] = ap[16 * half + 8 + i];
    }
    const bf16_t* bp = Bt + (size_t)(n0 + r) * ldb + kb + 16 * half;
#pragma unroll
    for (int i = 0; i < 16; ++i) b.e[i] = bp[i];
    acc = __builtin_amdgcn_wmma_f32_16x16x32_bf16(false, a.v, false, b.v,
                                                  (short)0, acc, false, false);
  }
  return acc;
}

// pre-transpose weights into bf16 [Ncols][K] layout in workspace
__global__ void __launch_bounds__(256)
prep_weights(const float* __restrict__ Wl, const float* __restrict__ Wr,
             const float* __restrict__ W1, const float* __restrict__ W2,
             bf16_t* __restrict__ ws) {
  int i = blockIdx.x * blockDim.x + threadIdx.x;
  if (i < 65536) {                 // WlT[n][k] = Wl[k][n], n<512,k<128
    int n = i >> 7, k = i & 127;
    ws[i] = (bf16_t)Wl[k * 512 + n];
  } else if (i < 131072) {
    int j = i - 65536; int n = j >> 7, k = j & 127;
    ws[i] = (bf16_t)Wr[k * 512 + n];
  } else if (i < 147456) {
    int j = i - 131072; int n = j >> 7, k = j & 127;
    ws[i] = (bf16_t)W1[k * 128 + n];
  } else if (i < 163840) {
    int j = i - 147456; int n = j >> 7, k = j & 127;
    ws[i] = (bf16_t)W2[k * 128 + n];
  }
}

__global__ void __launch_bounds__(256)
gat_fused(const float* __restrict__ qemb, const int* __restrict__ qmask,
          const int* __restrict__ eidx, const int* __restrict__ ew,
          const int* __restrict__ emki,
          const float* __restrict__ ln1g, const float* __restrict__ ln1b,
          const float* __restrict__ embt, const float* __restrict__ We,
          const float* __restrict__ bl, const float* __restrict__ br,
          const float* __restrict__ att, const float* __restrict__ gatb,
          const float* __restrict__ b1, const float* __restrict__ b2,
          const float* __restrict__ ln2g, const float* __restrict__ ln2b,
          const bf16_t* __restrict__ WlT, const bf16_t* __restrict__ WrT,
          const bf16_t* __restrict__ W1T, const bf16_t* __restrict__ W2T,
          float* __restrict__ out) {
  extern __shared__ char smem[];
  float*  xl     = (float*)(smem + OFF_XL);
  float*  xr     = (float*)(smem + OFF_XR);
  float*  We_s   = (float*)(smem + OFF_WE);
  float*  att_s  = (float*)(smem + OFF_ATT);
  bf16_t* xbf    = (bf16_t*)(smem + OFF_XBF);
  bf16_t* h1bf   = (bf16_t*)(smem + OFF_H1BF);
  float*  eembs  = (float*)(smem + OFF_EEMB);
  int*    srcs   = (int*)(smem + OFF_SRC);
  int*    dsts   = (int*)(smem + OFF_DST);
  float*  emk    = (float*)(smem + OFF_EMK);
  float*  loopat = (float*)(smem + OFF_LOOP);
  float*  ecnt   = (float*)(smem + OFF_ECNT);
  float*  lg     = (float*)(smem + OFF_LOG);
  float*  lgs    = (float*)(smem + OFF_LGS);
  float*  mS     = (float*)(smem + OFF_M);
  float*  denS   = (float*)(smem + OFF_DEN);
  float*  pe     = (float*)(smem + OFF_PE);
  float*  als    = (float*)(smem + OFF_ALS);
  float*  red    = (float*)(smem + OFF_RED);
  float*  pooled = (float*)(smem + OFF_POOL);
  float*  stats  = (float*)(smem + OFF_STAT);

  const int g = blockIdx.x;
  const int tid = threadIdx.x;
  const int lane = tid & 31;
  const int wave = tid >> 5;

  // ---- stage small tensors + edge lists ----
  for (int i = tid; i < 4096; i += 256) We_s[i] = We[i];
  for (int i = tid; i < 512; i += 256) att_s[i] = att[i];
  if (tid < Ee) {
    srcs[tid] = eidx[g * 128 + tid] & 31;
    dsts[tid] = eidx[g * 128 + 64 + tid] & 31;
    emk[tid] = (float)emki[g * 64 + tid];
    int w = ew[g * 64 + tid] & 1;
    for (int c = 0; c < 8; ++c) eembs[tid * 8 + c] = embt[w * 8 + c];
  }
  if (tid < Nn) {
    ecnt[tid] = 0.f;
    for (int c = 0; c < 8; ++c) loopat[tid * 8 + c] = 0.f;
  }
  __syncthreads();

  // ---- edge stats (LDS atomics) + LN1 -> bf16 tile ----
  if (tid < Ee) {
    float m = emk[tid]; int d = dsts[tid];
    atomicAdd(&ecnt[d], m);
    for (int c = 0; c < 8; ++c) atomicAdd(&loopat[d * 8 + c], m * eembs[tid * 8 + c]);
  }
  {
    const int row = tid >> 3, t8 = tid & 7;
    const float* xg = qemb + ((size_t)g * 32 + row) * 128 + t8 * 16;
    float s = 0.f, ss = 0.f, v[16];
#pragma unroll
    for (int i = 0; i < 16; ++i) { v[i] = xg[i]; s += v[i]; ss += v[i] * v[i]; }
    red[tid] = s; red[256 + tid] = ss;
    __syncthreads();
    float S = 0.f, SS = 0.f;
#pragma unroll
    for (int j = 0; j < 8; ++j) { S += red[row * 8 + j]; SS += red[256 + row * 8 + j]; }
    float mu = S * (1.f / 128.f);
    float rinv = rsqrtf(SS * (1.f / 128.f) - mu * mu + 1e-5f);
#pragma unroll
    for (int i = 0; i < 16; ++i) {
      int k = t8 * 16 + i;
      xbf[row * 128 + k] = (bf16_t)((v[i] - mu) * rinv * ln1g[k] + ln1b[k]);
    }
  }
  __syncthreads();

  // normalize self-loop edge attrs
  { int n = tid >> 3, c = tid & 7; loopat[n * 8 + c] /= fmaxf(ecnt[n], 1.f); }

  // ---- GEMM1: xl / xr = LN(x) @ {Wl,Wr} + bias  (128 16x16 tiles, WMMA bf16) ----
  for (int t = wave; t < 128; t += 8) {
    int mt = t & 1, nt = t >> 1;
    float* outp; const bf16_t* Bt; const float* bias; int n0;
    if (nt < 32) { outp = xl; Bt = WlT; bias = bl; n0 = nt * 16; }
    else         { outp = xr; Bt = WrT; bias = br; n0 = (nt - 32) * 16; }
    CF acc; acc.v = wmma_tile128(xbf, 128, mt * 16, Bt, 128, n0, lane);
    int r = lane & 15, half = lane >> 4;
    float bv = bias[n0 + r];
#pragma unroll
    for (int i = 0; i < 8; ++i)
      outp[(mt * 16 + i + 8 * half) * LDH + n0 + r] = acc.f[i] + bv;
  }
  __syncthreads();

  // ---- edge / self-loop attention logits ----
  {
    int e = tid >> 2, h = tid & 3;
    const float* xls = xl + srcs[e] * LDH + h * 128;
    const float* xrd = xr + dsts[e] * LDH + h * 128;
    const float* ee = eembs + e * 8;
    float acc = 0.f;
    for (int d = 0; d < 128; ++d) {
      float ev = 0.f;
#pragma unroll
      for (int c = 0; c < 8; ++c) ev += ee[c] * We_s[c * 512 + h * 128 + d];
      float z = xls[d] + xrd[d] + ev;
      z = z > 0.f ? z : 0.2f * z;
      acc += att_s[h * 128 + d] * z;
    }
    lg[tid] = emk[e] > 0.f ? acc : -1e30f;
  }
  if (tid < 128) {
    int n = tid >> 2, h = tid & 3;
    const float* la = loopat + n * 8;
    float acc = 0.f;
    for (int d = 0; d < 128; ++d) {
      float ev = 0.f;
#pragma unroll
      for (int c = 0; c < 8; ++c) ev += la[c] * We_s[c * 512 + h * 128 + d];
      float z = xl[n * LDH + h * 128 + d] + xr[n * LDH + h * 128 + d] + ev;
      z = z > 0.f ? z : 0.2f * z;
      acc += att_s[h * 128 + d] * z;
    }
    lgs[tid] = acc;
  }
  __syncthreads();

  // ---- segment softmax ----
  if (tid < 128) {
    int n = tid >> 2, h = tid & 3;
    float mm = lgs[tid];
    for (int e = 0; e < 64; ++e) if (dsts[e] == n) mm = fmaxf(mm, lg[e * 4 + h]);
    mS[tid] = mm;
  }
  __syncthreads();
  {
    int e = tid >> 2, h = tid & 3;
    pe[tid] = emk[e] > 0.f ? __expf(lg[tid] - mS[dsts[e] * 4 + h]) : 0.f;
  }
  __syncthreads();
  if (tid < 128) {
    int n = tid >> 2, h = tid & 3;
    float es = __expf(lgs[tid] - mS[tid]);
    float den = es;
    for (int e = 0; e < 64; ++e) if (dsts[e] == n) den += pe[e * 4 + h];
    denS[tid] = den;
    als[tid] = es / den;
  }
  __syncthreads();

  // ---- aggregation: agg aliases xr (dead after logits) ----
  float* agg = xr;
  for (int idx = tid; idx < 32 * 512; idx += 256) {
    int n = idx >> 9, col = idx & 511, h = col >> 7;
    agg[n * LDH + col] = als[n * 4 + h] * xl[n * LDH + col];
  }
  __syncthreads();
  for (int idx = tid; idx < 64 * 512; idx += 256) {
    int e = idx >> 9, col = idx & 511, h = col >> 7;
    float a = pe[e * 4 + h];
    if (a != 0.f) {
      a /= denS[dsts[e] * 4 + h];
      atomicAdd(&agg[dsts[e] * LDH + col], a * xl[srcs[e] * LDH + col]);
    }
  }
  __syncthreads();

  // ---- head mean + gat bias -> bf16 (aliases xbf) ----
  bf16_t* goutb = xbf;
  for (int idx = tid; idx < 4096; idx += 256) {
    int n = idx >> 7, d = idx & 127;
    float v = 0.25f * (agg[n * LDH + d] + agg[n * LDH + 128 + d] +
                       agg[n * LDH + 256 + d] + agg[n * LDH + 384 + d]) + gatb[d];
    goutb[idx] = (bf16_t)v;
  }
  __syncthreads();

  // ---- FFN GEMM1 (leaky 0.01) ----
  for (int t = wave; t < 16; t += 8) {
    int mt = t & 1, n0 = (t >> 1) * 16;
    CF acc; acc.v = wmma_tile128(goutb, 128, mt * 16, W1T, 128, n0, lane);
    int r = lane & 15, half = lane >> 4;
    float bv = b1[n0 + r];
#pragma unroll
    for (int i = 0; i < 8; ++i) {
      float z = acc.f[i] + bv;
      z = z > 0.f ? z : 0.01f * z;
      h1bf[(mt * 16 + i + 8 * half) * 128 + n0 + r] = (bf16_t)z;
    }
  }
  __syncthreads();

  // ---- FFN GEMM2 (hout aliases xl) ----
  float* hout = xl;
  for (int t = wave; t < 16; t += 8) {
    int mt = t & 1, n0 = (t >> 1) * 16;
    CF acc; acc.v = wmma_tile128(h1bf, 128, mt * 16, W2T, 128, n0, lane);
    int r = lane & 15, half = lane >> 4;
    float bv = b2[n0 + r];
#pragma unroll
    for (int i = 0; i < 8; ++i)
      hout[(mt * 16 + i + 8 * half) * 128 + n0 + r] = acc.f[i] + bv;
  }
  __syncthreads();

  // ---- masked mean pool + valid gate ----
  if (tid == 0) {
    float ns = 0.f; for (int n = 0; n < 32; ++n) ns += (float)qmask[g * 32 + n];
    float es = 0.f; for (int e = 0; e < 64; ++e) es += emk[e];
    float valid = (ns > 0.f && es > 0.f) ? 1.f : 0.f;
    stats[0] = valid / fmaxf(ns, 1.f);
  }
  __syncthreads();
  if (tid < 128) {
    float p = 0.f;
    for (int n = 0; n < 32; ++n) p += hout[n * 128 + tid] * (float)qmask[g * 32 + n];
    pooled[tid] = p * stats[0];
  }
  __syncthreads();

  // ---- LN2 + store ----
  if (tid == 0) {
    float S = 0.f, SS = 0.f;
    for (int d = 0; d < 128; ++d) { float v = pooled[d]; S += v; SS += v * v; }
    float mu = S * (1.f / 128.f);
    stats[1] = mu;
    stats[2] = rsqrtf(SS * (1.f / 128.f) - mu * mu + 1e-5f);
  }
  __syncthreads();
  if (tid < 128)
    out[(size_t)g * 128 + tid] =
        (pooled[tid] - stats[1]) * stats[2] * ln2g[tid] + ln2b[tid];
}

extern "C" void kernel_launch(void* const* d_in, const int* in_sizes, int n_in,
                              void* d_out, int out_size, void* d_ws, size_t ws_size,
                              hipStream_t stream) {
  (void)in_sizes; (void)n_in; (void)out_size; (void)ws_size;
  const float* qemb = (const float*)d_in[0];
  const int*   qm   = (const int*)d_in[1];
  const int*   eix  = (const int*)d_in[2];
  const int*   ewt  = (const int*)d_in[3];
  const int*   emm  = (const int*)d_in[4];
  const float* l1g  = (const float*)d_in[5];
  const float* l1b  = (const float*)d_in[6];
  const float* emt  = (const float*)d_in[7];
  const float* Wl   = (const float*)d_in[8];
  const float* bl   = (const float*)d_in[9];
  const float* Wr   = (const float*)d_in[10];
  const float* br   = (const float*)d_in[11];
  const float* We   = (const float*)d_in[12];
  const float* att  = (const float*)d_in[13];
  const float* gb   = (const float*)d_in[14];
  const float* W1   = (const float*)d_in[15];
  const float* b1   = (const float*)d_in[16];
  const float* W2   = (const float*)d_in[17];
  const float* b2   = (const float*)d_in[18];
  const float* l2g  = (const float*)d_in[19];
  const float* l2b  = (const float*)d_in[20];
  float* out = (float*)d_out;
  bf16_t* wt = (bf16_t*)d_ws;

  prep_weights<<<640, 256, 0, stream>>>(Wl, Wr, W1, W2, wt);

  hipFuncSetAttribute((const void*)gat_fused,
                      hipFuncAttributeMaxDynamicSharedMemorySize, SMEM_BYTES);
  gat_fused<<<Gc, 256, SMEM_BYTES, stream>>>(
      qemb, qm, eix, ewt, emm, l1g, l1b, emt, We, bl, br, att, gb, b1, b2,
      l2g, l2b, wt, wt + 65536, wt + 131072, wt + 147456, out);
}
